// MultiHeadAttention_18511309045773
// MI455X (gfx1250) — compile-verified, matrix operand order checked by reading, not run on hardware
//
#include <hip/hip_runtime.h>
#include <hip/hip_bf16.h>

#define HIDDEN  1024
#define NHEADS  16
#define HEADDIM 64
#define BATCH   4
#define SEQ     2048
#define MROWS   (BATCH * SEQ)   // 8192

typedef __attribute__((ext_vector_type(16))) __bf16   v16bf;
typedef __attribute__((ext_vector_type(8)))  float    v8f;
typedef __attribute__((ext_vector_type(4)))  unsigned uv4;
typedef __attribute__((ext_vector_type(4)))  float    fv4;
typedef __attribute__((ext_vector_type(4)))  unsigned tdm_uv4;
typedef __attribute__((ext_vector_type(8)))  int      tdm_iv8;
typedef __attribute__((ext_vector_type(4)))  int      tdm_iv4;

#if defined(__gfx1250__) && __has_builtin(__builtin_amdgcn_tensor_load_to_lds)
#define USE_TDM 1
#endif

#define WMMA_BF16(A, B, C) \
  __builtin_amdgcn_wmma_f32_16x16x32_bf16(false, (A), false, (B), (short)0, (C), false, false)

// 16x32 bf16 operand fragment: per-lane 16 values in 8 dwords.
// Lanes 0-15 hold K 0..7 / 16..23; lanes 16-31 hold K 8..15 / 24..31.
union Frag { uv4 q[2]; v16bf v; };

__device__ __forceinline__ unsigned short f2bf(float f) {
  unsigned u = __float_as_uint(f);
  u += 0x7FFFu + ((u >> 16) & 1u);          // round-to-nearest-even
  return (unsigned short)(u >> 16);
}
__device__ __forceinline__ unsigned pk2(float a, float b) {
  return (unsigned)f2bf(a) | ((unsigned)f2bf(b) << 16);
}
__device__ __forceinline__ void load_frag(Frag& f, const unsigned short* rowBase, int half) {
  const unsigned short* p = rowBase + half * 8;
  f.q[0] = *(const uv4*)(p);
  f.q[1] = *(const uv4*)(p + 16);
}

// ---------------------------------------------------------------------------
// Kernel 1a: W[k][n] f32 -> Wt[n][k] bf16  (B operand becomes contiguous)
// ---------------------------------------------------------------------------
__global__ __launch_bounds__(256) void transpose_bf16(const float* __restrict__ W,
                                                      unsigned short* __restrict__ Wt) {
  int idx = blockIdx.x * 256 + threadIdx.x;     // 0 .. 1M-1
  int k = idx >> 10, n = idx & (HIDDEN - 1);
  Wt[(size_t)n * HIDDEN + k] = f2bf(W[idx]);
}

// ---------------------------------------------------------------------------
// Kernel 1b: streaming f32 -> bf16 convert (activations), 8 elems/thread
// ---------------------------------------------------------------------------
__global__ __launch_bounds__(256) void convert_bf16(const float* __restrict__ X,
                                                    unsigned short* __restrict__ Y) {
  int i = (blockIdx.x * 256 + threadIdx.x) * 8;
  fv4 a = *(const fv4*)(X + i);
  fv4 b = *(const fv4*)(X + i + 4);
  uv4 o;
  o.x = pk2(a.x, a.y); o.y = pk2(a.z, a.w);
  o.z = pk2(b.x, b.y); o.w = pk2(b.z, b.w);
  *(uv4*)(Y + i) = o;
}

// ---------------------------------------------------------------------------
// TDM: DMA a (rows x 32) bf16 tile (row stride 1024 elems) into LDS.
// D# built per CDNA5 ISA 8.3/8.4: group0 = count|lds_addr|global_addr|type=2,
// group1 = data_size=2B, tensor_dim0=1024, tile_dim0=32, tile_dim1=128,
// tensor_dim0_stride=1024.  2D tensor -> groups 2/3 (and spare group) zero.
// ---------------------------------------------------------------------------
#if defined(USE_TDM)
__device__ __forceinline__ void tdm_load_tile(const unsigned short* gsrc,
                                              unsigned ldsByteAddr,
                                              unsigned tensorDim1) {
  unsigned long long ga = (unsigned long long)(size_t)gsrc;
  tdm_uv4 g0;
  g0.x = 1u;                                        // count=1, user mode
  g0.y = ldsByteAddr;                               // lds_addr [63:32]
  g0.z = (unsigned)ga;                              // global_addr low
  g0.w = (unsigned)((ga >> 32) & 0x01FFFFFFu) | 0x80000000u;  // addr hi | type=2
  tdm_iv8 g1;
  g1[0] = 0x00010000;                               // data_size=1 (2 bytes)
  g1[1] = (int)(1024u << 16);                       // tensor_dim0=1024 (low16 @ [63:48])
  g1[2] = (int)((tensorDim1 & 0xFFFFu) << 16);      // dim0 hi16=0 | tensor_dim1 low16
  g1[3] = (int)(((tensorDim1 >> 16) & 0xFFFFu) | (32u << 16)); // dim1 hi | tile_dim0=32
  g1[4] = 128;                                      // tile_dim1=128, tile_dim2=0
  g1[5] = 1024;                                     // tensor_dim0_stride low32
  g1[6] = 0;                                        // stride hi | dim1_stride low
  g1[7] = 0;
  tdm_iv4 z4 = {0, 0, 0, 0};
  tdm_iv8 z8 = {0, 0, 0, 0, 0, 0, 0, 0};
  __builtin_amdgcn_tensor_load_to_lds(g0, g1, z4, z4, z8, 0);
}
#endif

// ---------------------------------------------------------------------------
// Kernel 2: GEMM  out[m][n] = A[m][:] . Wt[n][:] + bias[n]   (all-bf16 A/W)
//   128x128 block tile, 256 threads = 8 waves (4x2), wave tile 32x64 (2x4 WMMA)
//   mode 0: bf16 [B,H,S,D] (Q,K) | mode 1: bf16 [B,H,D,S] (V^T) | mode 2: f32+bias
// ---------------------------------------------------------------------------
__global__ __launch_bounds__(256)
void gemm_proj(const unsigned short* __restrict__ Abf,
               const unsigned short* __restrict__ Wt,
               const float* __restrict__ bias,
               unsigned short* __restrict__ out16,
               float* __restrict__ out32, int mode) {
  __shared__ __align__(16) unsigned short sX[128 * 32];
  __shared__ __align__(16) unsigned short sW[128 * 32];

  const int tid  = threadIdx.x;
  const int lane = tid & 31;
  const int l16  = lane & 15;
  const int half = lane >> 4;
  const int wv   = tid >> 5;
  const int wm   = (wv >> 1) * 32;   // 0,32,64,96
  const int wn   = (wv & 1) * 64;    // 0,64
  const int blockN = blockIdx.x * 128;
  const int blockM = blockIdx.y * 128;

  v8f acc[2][4];
#pragma unroll
  for (int i = 0; i < 2; ++i)
#pragma unroll
    for (int j = 0; j < 4; ++j) acc[i][j] = {};

#if defined(USE_TDM)
  const unsigned ldsX = (unsigned)(size_t)(sX);
  const unsigned ldsW = (unsigned)(size_t)(sW);
#else
  const int ldRow = tid >> 1;            // 0..127
  const int ldSeg = (tid & 1) * 16;      // 0 or 16
#endif

  for (int k0 = 0; k0 < HIDDEN; k0 += 32) {
#if defined(USE_TDM)
    if (wv == 0) {                       // one wave drives both tile DMAs
      tdm_load_tile(Abf + (size_t)blockM * HIDDEN + k0, ldsX, MROWS);
      tdm_load_tile(Wt  + (size_t)blockN * HIDDEN + k0, ldsW, HIDDEN);
      __builtin_amdgcn_s_wait_tensorcnt(0);
    }
#else
    {
      const unsigned short* g = Abf + (size_t)(blockM + ldRow) * HIDDEN + k0 + ldSeg;
      uv4 a = *(const uv4*)g;
      uv4 b = *(const uv4*)(g + 8);
      *(uv4*)(sX + ldRow * 32 + ldSeg)     = a;
      *(uv4*)(sX + ldRow * 32 + ldSeg + 8) = b;
    }
    {
      const unsigned short* g = Wt + (size_t)(blockN + ldRow) * HIDDEN + k0 + ldSeg;
      if (k0 + 32 < HIDDEN) __builtin_prefetch(g + 32, 0, 0);
      uv4 a = *(const uv4*)g;
      uv4 b = *(const uv4*)(g + 8);
      *(uv4*)(sW + ldRow * 32 + ldSeg)     = a;
      *(uv4*)(sW + ldRow * 32 + ldSeg + 8) = b;
    }
#endif
    __syncthreads();

    Frag aF[2], bF[4];
#pragma unroll
    for (int i = 0; i < 2; ++i)
      load_frag(aF[i], sX + (wm + i * 16 + l16) * 32, half);
#pragma unroll
    for (int j = 0; j < 4; ++j)
      load_frag(bF[j], sW + (wn + j * 16 + l16) * 32, half);

#pragma unroll
    for (int i = 0; i < 2; ++i)
#pragma unroll
      for (int j = 0; j < 4; ++j)
        acc[i][j] = WMMA_BF16(aF[i].v, bF[j].v, acc[i][j]);
    __syncthreads();
  }

  // epilogue: C layout = row in VGPR index (+8 for lanes 16-31), col in lane
#pragma unroll
  for (int i = 0; i < 2; ++i) {
#pragma unroll
    for (int j = 0; j < 4; ++j) {
      const int col = blockN + wn + j * 16 + l16;
      const float bc = bias[col];
#pragma unroll
      for (int v = 0; v < 8; ++v) {
        const int m = blockM + wm + i * 16 + v + half * 8;
        const float val = acc[i][j][v] + bc;
        if (mode == 2) {
          out32[(size_t)m * HIDDEN + col] = val;
        } else {
          const int b = m >> 11, s = m & (SEQ - 1);
          const int h = col >> 6, d = col & (HEADDIM - 1);
          const int bh = b * NHEADS + h;
          if (mode == 0)
            out16[((size_t)bh * SEQ + s) * HEADDIM + d] = f2bf(val);
          else
            out16[((size_t)bh * HEADDIM + d) * SEQ + s] = f2bf(val);
        }
      }
    }
  }
}

// ---------------------------------------------------------------------------
// Kernel 3: flash attention. One wave per (16-query tile, b*h).
// ---------------------------------------------------------------------------
__global__ __launch_bounds__(32)
void flash_attn(const unsigned short* __restrict__ Qbf,
                const unsigned short* __restrict__ Kbf,
                const unsigned short* __restrict__ Vt,
                unsigned short* __restrict__ Ctx) {
  __shared__ __align__(16) unsigned short sP[16 * 32];

  const int lane = threadIdx.x;
  const int l16  = lane & 15;
  const int half = lane >> 4;
  const int qt = blockIdx.x;
  const int bh = blockIdx.y;
  const int b = bh >> 4, h = bh & (NHEADS - 1);

  Frag qf[2];
  {
    const unsigned short* qrow = Qbf + ((size_t)bh * SEQ + qt * 16 + l16) * HEADDIM;
#pragma unroll
    for (int f = 0; f < 2; ++f) load_frag(qf[f], qrow + f * 32, half);
  }

  v8f o[4]; o[0] = {}; o[1] = {}; o[2] = {}; o[3] = {};
  float ml[8], ll[8];
#pragma unroll
  for (int v = 0; v < 8; ++v) { ml[v] = -1e30f; ll[v] = 0.0f; }

  for (int chunk = 0; chunk < SEQ / 32; ++chunk) {
    v8f c[2];
#pragma unroll
    for (int ct = 0; ct < 2; ++ct) {
      Frag k0, k1;
      const unsigned short* krow =
          Kbf + ((size_t)bh * SEQ + chunk * 32 + ct * 16 + l16) * HEADDIM;
      load_frag(k0, krow, half);
      load_frag(k1, krow + 32, half);
      v8f z = {};
      z = WMMA_BF16(qf[0].v, k0.v, z);
      z = WMMA_BF16(qf[1].v, k1.v, z);
      c[ct] = z * 0.125f;                       // 1/sqrt(64)
    }

    // online softmax: rows in VGPR index, cols across 16-lane halves
#pragma unroll
    for (int v = 0; v < 8; ++v) {
      float a0 = c[0][v], a1 = c[1][v];
      float mx = fmaxf(a0, a1);
      mx = fmaxf(mx, __shfl_xor(mx, 1, 32));
      mx = fmaxf(mx, __shfl_xor(mx, 2, 32));
      mx = fmaxf(mx, __shfl_xor(mx, 4, 32));
      mx = fmaxf(mx, __shfl_xor(mx, 8, 32));
      const float mnew  = fmaxf(ml[v], mx);
      const float alpha = __expf(ml[v] - mnew);
      const float p0 = __expf(a0 - mnew);
      const float p1 = __expf(a1 - mnew);
      float rs = p0 + p1;
      rs += __shfl_xor(rs, 1, 32);
      rs += __shfl_xor(rs, 2, 32);
      rs += __shfl_xor(rs, 4, 32);
      rs += __shfl_xor(rs, 8, 32);
      ll[v] = ll[v] * alpha + rs;
      ml[v] = mnew;
#pragma unroll
      for (int dt = 0; dt < 4; ++dt) o[dt][v] *= alpha;
      const int r = v + half * 8;
      sP[r * 32 + l16]      = f2bf(p0);
      sP[r * 32 + 16 + l16] = f2bf(p1);
    }
    __syncthreads();                            // 1-wave block: lowers to S_NOP

    Frag pf;
    load_frag(pf, sP + l16 * 32, half);         // C-layout -> A-layout via LDS

#pragma unroll
    for (int dt = 0; dt < 4; ++dt) {
      Frag vf;
      const unsigned short* vrow =
          Vt + ((size_t)bh * HEADDIM + dt * 16 + l16) * SEQ + chunk * 32;
      load_frag(vf, vrow, half);
      o[dt] = WMMA_BF16(pf.v, vf.v, o[dt]);
    }
    __syncthreads();
  }

#pragma unroll
  for (int v = 0; v < 8; ++v) {
    const float inv = 1.0f / ll[v];
    const int s = qt * 16 + v + half * 8;
#pragma unroll
    for (int dt = 0; dt < 4; ++dt) {
      const int col = h * HEADDIM + dt * 16 + l16;
      Ctx[((size_t)(b * SEQ + s)) * HIDDEN + col] = f2bf(o[dt][v] * inv);
    }
  }
}

// ---------------------------------------------------------------------------
// Launch.  Workspace (72 MB) with lifetime-based aliasing:
//   [0,8)MB   WqT,WkT,WvT,WoT
//   [8,24)    Xq  -> later Kbf      (Xq dead after GEMM-Q)
//   [24,40)   Xk  -> later Vt       (Xk dead after GEMM-K)
//   [40,56)   Xv  -> later Ctx      (Xv dead after GEMM-V)
//   [56,72)   Qbf
// ---------------------------------------------------------------------------
extern "C" void kernel_launch(void* const* d_in, const int* in_sizes, int n_in,
                              void* d_out, int out_size, void* d_ws, size_t ws_size,
                              hipStream_t stream) {
  (void)in_sizes; (void)n_in; (void)out_size; (void)ws_size;

  const float* query = (const float*)d_in[0];
  const float* key   = (const float*)d_in[1];
  const float* value = (const float*)d_in[2];
  const float* Wq = (const float*)d_in[3];
  const float* bq = (const float*)d_in[4];
  const float* Wk = (const float*)d_in[5];
  const float* bk = (const float*)d_in[6];
  const float* Wv = (const float*)d_in[7];
  const float* bv = (const float*)d_in[8];
  const float* Wo = (const float*)d_in[9];
  const float* bo = (const float*)d_in[10];

  char* w = (char*)d_ws;
  const size_t MB = 1u << 20;
  unsigned short* WqT = (unsigned short*)(w + 0 * MB);
  unsigned short* WkT = (unsigned short*)(w + 2 * MB);
  unsigned short* WvT = (unsigned short*)(w + 4 * MB);
  unsigned short* WoT = (unsigned short*)(w + 6 * MB);
  unsigned short* Xq  = (unsigned short*)(w + 8 * MB);
  unsigned short* Xk  = (unsigned short*)(w + 24 * MB);
  unsigned short* Xv  = (unsigned short*)(w + 40 * MB);
  unsigned short* Qbf = (unsigned short*)(w + 56 * MB);
  unsigned short* Kbf = Xq;   // aliases (see lifetimes above)
  unsigned short* Vtb = Xk;
  unsigned short* Ctx = Xv;

  transpose_bf16<<<(HIDDEN * HIDDEN) / 256, 256, 0, stream>>>(Wq, WqT);
  transpose_bf16<<<(HIDDEN * HIDDEN) / 256, 256, 0, stream>>>(Wk, WkT);
  transpose_bf16<<<(HIDDEN * HIDDEN) / 256, 256, 0, stream>>>(Wv, WvT);
  transpose_bf16<<<(HIDDEN * HIDDEN) / 256, 256, 0, stream>>>(Wo, WoT);

  const int convBlocks = (MROWS * HIDDEN) / (256 * 8);   // 4096
  convert_bf16<<<convBlocks, 256, 0, stream>>>(query, Xq);
  convert_bf16<<<convBlocks, 256, 0, stream>>>(key,   Xk);
  convert_bf16<<<convBlocks, 256, 0, stream>>>(value, Xv);

  dim3 gg(HIDDEN / 128, MROWS / 128);   // (8, 64)
  gemm_proj<<<gg, 256, 0, stream>>>(Xq,  WqT, bq, Qbf, nullptr, 0);
  gemm_proj<<<gg, 256, 0, stream>>>(Xk,  WkT, bk, Kbf, nullptr, 0);
  gemm_proj<<<gg, 256, 0, stream>>>(Xv,  WvT, bv, Vtb, nullptr, 1);

  flash_attn<<<dim3(SEQ / 16, BATCH * NHEADS), 32, 0, stream>>>(Qbf, Kbf, Vtb, Ctx);

  gemm_proj<<<gg, 256, 0, stream>>>(Ctx, WoT, bo, nullptr, (float*)d_out, 2);
}
